// SALAD_18846316495042
// MI455X (gfx1250) — compile-verified, matrix operand order checked by reading, not run on hardware
//
#include <hip/hip_runtime.h>
#include <math.h>

#define B_   32
#define N_   1025
#define D_   768
#define C_   64
#define CD_  128
#define TD_  256
#define NTOK 1024
#define SINKHORN 5

#define K1_TILES  12   // 8 tiles of f-columns (128) + 4 tiles of s-columns (64)
#define K1_KSTEPS 24   // 768 / 32

typedef __attribute__((ext_vector_type(16))) __bf16         v16bf;
typedef __attribute__((ext_vector_type(2)))  __bf16         v2bf;
typedef __attribute__((ext_vector_type(16))) unsigned short v16u;
typedef __attribute__((ext_vector_type(8)))  float          v8f;

union FragU { v16u u; v16bf b; uint4 q[2]; unsigned d[8]; };

// Packed f32x2 -> bf16x2: one v_cvt_pk_bf16_f32 (verified selected on gfx1250).
__device__ __forceinline__ unsigned cvt_pk(float lo, float hi) {
#if __has_builtin(__builtin_amdgcn_cvt_pk_bf16_f32)
  union { v2bf v; unsigned u; } c;
  c.v = __builtin_amdgcn_cvt_pk_bf16_f32(lo, hi);
  return c.u;
#else
  union { v2bf v; unsigned u; } c;
  c.v[0] = (__bf16)lo; c.v[1] = (__bf16)hi;
  return c.u;
#endif
}

// Generic pointer -> raw LDS byte offset (addrspacecast to AS(3)).
__device__ __forceinline__ unsigned lds_off(const void* p) {
  return (unsigned)(unsigned long long)(__attribute__((address_space(3))) const void*)p;
}

// K index within a 32-wide K step for element e of a 16-element A/B fragment.
// Layout (ISA 7.12.2, 16-bit A 16x32): lanes 0-15 hold K[0..7],K[16..23];
// lanes 16-31 hold K[8..15],K[24..31].
__device__ __forceinline__ int kidx(int e, int khalf) {
  return (e < 8) ? (khalf * 8 + e) : (16 + khalf * 8 + (e - 8));
}

__device__ __forceinline__ void cvt_a(FragU& a, const float4& r0, const float4& r1,
                                      const float4& r2, const float4& r3) {
  a.d[0] = cvt_pk(r0.x, r0.y); a.d[1] = cvt_pk(r0.z, r0.w);
  a.d[2] = cvt_pk(r1.x, r1.y); a.d[3] = cvt_pk(r1.z, r1.w);
  a.d[4] = cvt_pk(r2.x, r2.y); a.d[5] = cvt_pk(r2.z, r2.w);
  a.d[6] = cvt_pk(r3.x, r3.y); a.d[7] = cvt_pk(r3.z, r3.w);
}

// Build an A fragment from a row-major f32 row: run of 8 at p, run of 8 at p+16.
__device__ __forceinline__ v16bf make_a_frag(const float* p) {
  float4 r0 = *(const float4*)(p);
  float4 r1 = *(const float4*)(p + 4);
  float4 r2 = *(const float4*)(p + 16);
  float4 r3 = *(const float4*)(p + 20);
  FragU f; cvt_a(f, r0, r1, r2, r3);
  return f.b;
}

// ---------------------------------------------------------------------------
// Kernel 1: one streaming pass over fi = x[:,1:,:], producing
//   fbufq : f = fi@Wf + bf, stored as pre-swizzled bf16 WMMA B-fragments
//   Mbuf[b, c, n] = (fi@Ws + bs) * sharpness / REG   (c < 64), f32
// Weights live in LDS as pre-swizzled bf16 B-fragments, K-step-major so all 12
// fragments of a K-step are contiguous (12 KB) -> immediate DS offsets only.
// ---------------------------------------------------------------------------
__global__ void __launch_bounds__(256)
k_fused_gemm(const float* __restrict__ x, const float* __restrict__ Wf,
             const float* __restrict__ bf, const float* __restrict__ Ws,
             const float* __restrict__ bs, const float* __restrict__ sharpness,
             unsigned* __restrict__ fbufq, float* __restrict__ Mbuf)
{
  extern __shared__ unsigned wlds[];   // 24*12*256 dwords = 288 KB
  const int tid = threadIdx.x;

  // Stage [Wf | Ws] into LDS, fragment index = ks*12 + ct.
  // Within a fragment: per lane 32B as 16B @ lane*16 and 16B @ 512+lane*16.
  for (int id = tid; id < K1_TILES * K1_KSTEPS * 256; id += 256) {
    int frag = id >> 8;
    int d    = id & 255;
    int ct   = frag % K1_TILES;
    int ks   = frag / K1_TILES;
    int h    = d >> 7;              // which 16B half
    int lane = (d & 127) >> 2;
    int j    = (h << 2) | (d & 3);  // dword 0..7 of the lane's 32B
    int khalf = lane >> 4;
    int c    = ct * 16 + (lane & 15);
    int e0   = 2 * j;
    int K0   = (j < 4) ? (khalf * 8 + e0) : (16 + khalf * 8 + (e0 - 8));
    int k0   = ks * 32 + K0;
    float w0, w1;
    if (c < CD_) { w0 = Wf[k0 * CD_ + c];        w1 = Wf[(k0 + 1) * CD_ + c]; }
    else         { int cs = c - CD_;
                   w0 = Ws[k0 * C_ + cs];        w1 = Ws[(k0 + 1) * C_ + cs]; }
    wlds[id] = cvt_pk(w0, w1);
  }
  __syncthreads();

  const int wave  = tid >> 5, lane = tid & 31;
  const int khalf = lane >> 4;
  const int rowbase = blockIdx.x * 128 + wave * 16;     // 16 rows per wave
  const int arow  = rowbase + (lane & 15);
  const int ab    = arow >> 10, an = arow & 1023;
  const float* arp = x + (size_t)ab * N_ * D_ + (size_t)(an + 1) * D_ + khalf * 8;
  const uint4* ldsq = (const uint4*)wlds;
  const float sh10 = sharpness[0] * 10.0f;              // sharpness / REG

  v8f acc[K1_TILES] = {};
  // Software-pipelined A fragment: raw loads for ks+1 issued before the 12
  // WMMAs of ks; converted afterwards.
  float4 r0, r1, r2, r3;
  FragU a;
  r0 = *(const float4*)(arp);      r1 = *(const float4*)(arp + 4);
  r2 = *(const float4*)(arp + 16); r3 = *(const float4*)(arp + 20);
  cvt_a(a, r0, r1, r2, r3);

  for (int ks = 0; ks < K1_KSTEPS; ++ks) {
    if (ks + 1 < K1_KSTEPS) {
      const float* ap = arp + (ks + 1) * 32;
      __builtin_prefetch(ap + 64, 0, 1);                // global_prefetch_b8
      r0 = *(const float4*)(ap);      r1 = *(const float4*)(ap + 4);
      r2 = *(const float4*)(ap + 16); r3 = *(const float4*)(ap + 20);
    }
    const uint4* bql = ldsq + (size_t)ks * (K1_TILES * 64) + lane;
    #pragma unroll
    for (int t = 0; t < K1_TILES; ++t) {
      FragU fb_;
      fb_.q[0] = bql[t * 64];                           // ds_load_b128 imm ofs
      fb_.q[1] = bql[t * 64 + 32];
      acc[t] = __builtin_amdgcn_wmma_f32_16x16x32_bf16(
                 false, a.b, false, fb_.b, (short)0, acc[t], false, false);
    }
    if (ks + 1 < K1_KSTEPS) cvt_a(a, r0, r1, r2, r3);
  }

  // Epilogue. C layout: lane holds col N=lane&15; vgpr vv holds row (lane/16)*8+vv.
  const int rb = khalf * 8;
  const int tokbase = rowbase + rb;          // 8 consecutive tokens, one batch
  const int bb2  = tokbase >> 10;
  const int nn0  = tokbase & 1023;
  const int ks2  = nn0 >> 5;                 // destination K-step in fbuf frags
  const int Kb   = nn0 & 31;                 // multiple of 8
  const int kh2  = (Kb >> 3) & 1;            // destination fragment lane-half
  const int eb   = (Kb >= 16) ? 8 : 0;       // destination element base
  const int lane2 = (lane & 15) + (kh2 << 4);

  #pragma unroll
  for (int t = 0; t < K1_TILES; ++t) {
    int col = t * 16 + (lane & 15);
    if (t < 8) {                                        // f-columns -> bf16 frags
      float bias = bf[col];
      unsigned* fragp = fbufq + (((size_t)(bb2 * 8 + t) * 32 + ks2) << 8);
      #pragma unroll
      for (int vv = 0; vv < 8; vv += 2) {
        int e = vv + eb;                                // even element index
        int j = e >> 1;
        unsigned dw = cvt_pk(acc[t][vv] + bias, acc[t][vv + 1] + bias);
        fragp[(j >> 2) * 128 + lane2 * 4 + (j & 3)] = dw;
      }
    } else {                                            // s-columns -> cost M (f32)
      int cs = col - CD_;
      float bias = bs[cs];
      #pragma unroll
      for (int vv = 0; vv < 8; ++vv) {
        int row = rowbase + rb + vv;
        int bb = row >> 10, nn = row & 1023;
        Mbuf[((size_t)bb * C_ + cs) * NTOK + nn] = (acc[t][vv] + bias) * sh10;
      }
    }
  }
}

// ---------------------------------------------------------------------------
// Kernel 2: per-batch LDS-resident Sinkhorn (65x1024 cost = 266 KB in LDS),
// then fused v_agg = 2*p@f - rowsum*anchors via bf16 WMMA.
// Cost matrix is pulled into LDS with async-to-LDS DMA (ASYNCcnt path).
// 1024 threads = 32 waves.
// ---------------------------------------------------------------------------
__global__ void __launch_bounds__(1024)
k_sinkhorn_vagg(const float* __restrict__ Mbuf, const unsigned* __restrict__ fbufq,
                const float* __restrict__ anchors, const float* __restrict__ dust_bin,
                float* __restrict__ out)
{
  extern __shared__ float sm2[];
  float* M      = sm2;                 // 65*1024
  float* u      = sm2 + 65 * NTOK;     // 80 (65 used)
  float* v      = u + 80;              // 1024
  float* rowsum = v + NTOK;            // 64
  const int b = blockIdx.x, tid = threadIdx.x;
  const int wave = tid >> 5, lane = tid & 31;

  // Async copy of cost rows 0..63 (256 KB) straight into LDS, no VGPR bounce.
  {
    const float* gsrc = Mbuf + (size_t)b * C_ * NTOK;
    for (int i = tid; i < (C_ * NTOK) / 4; i += 1024) {
      unsigned l = lds_off(&M[i * 4]);
      const float* g = gsrc + i * 4;
      asm volatile("global_load_async_to_lds_b128 %0, %1, off"
                   :: "v"(l), "v"(g) : "memory");
    }
    asm volatile("s_wait_asynccnt 0" ::: "memory");
  }
  M[C_ * NTOK + tid] = dust_bin[0] * 10.0f;   // dust row 64 = dust_bin / REG
  v[tid] = 0.0f;
  __syncthreads();

  const float LOGB    = -6.93147180559945f;     // -ln(1024)
  const float LA_DUST = -0.06453852113757f;     // ln(960/1024)

  for (int it = 0; it < SINKHORN; ++it) {
    // u = log_a - lse_n(M + v) : one wave per row, wave32 butterfly reduce
    for (int c = wave; c < C_ + 1; c += 32) {
      const float* Mr = M + c * NTOK;
      float mx = -3.0e38f;
      for (int n = lane; n < NTOK; n += 32) mx = fmaxf(mx, Mr[n] + v[n]);
      for (int off = 16; off; off >>= 1) mx = fmaxf(mx, __shfl_xor(mx, off, 32));
      float s = 0.0f;
      for (int n = lane; n < NTOK; n += 32) s += __expf(Mr[n] + v[n] - mx);
      for (int off = 16; off; off >>= 1) s += __shfl_xor(s, off, 32);
      float la = (c < C_) ? LOGB : LA_DUST;
      if (lane == 0) u[c] = la - (mx + __logf(s));
    }
    __syncthreads();
    // v = log_b - lse_c(M + u) : one thread per column (bank-conflict-free)
    {
      const int n = tid;
      float mx = -3.0e38f;
      for (int c = 0; c < C_ + 1; ++c) mx = fmaxf(mx, M[c * NTOK + n] + u[c]);
      float s = 0.0f;
      for (int c = 0; c < C_ + 1; ++c) s += __expf(M[c * NTOK + n] + u[c] - mx);
      v[n] = LOGB - (mx + __logf(s));
    }
    __syncthreads();
  }

  // p = exp(M + u + v) for rows 0..63, in place in LDS
  {
    const int n = tid; const float vn = v[n];
    for (int c = 0; c < C_; ++c)
      M[c * NTOK + n] = __expf(M[c * NTOK + n] + u[c] + vn);
  }
  __syncthreads();
  for (int c = wave; c < C_; c += 32) {
    const float* Mr = M + c * NTOK;
    float s = 0.0f;
    for (int n = lane; n < NTOK; n += 32) s += Mr[n];
    for (int off = 16; off; off >>= 1) s += __shfl_xor(s, off, 32);
    if (lane == 0) rowsum[c] = s;
  }
  __syncthreads();

  // v_agg: 64x128 output = 4x8 tiles, one 16x16 tile per wave, K = 1024.
  const int tr = wave >> 3, tc = wave & 7;
  const int khalf = lane >> 4;
  const int prow = tr * 16 + (lane & 15);
  const int ccol = tc * 16 + (lane & 15);
  const uint4* fq = (const uint4*)fbufq + (((size_t)(b * 8 + tc) * 32) << 6);
  v8f acc = {};
  FragU fcur;
  fcur.q[0] = fq[lane]; fcur.q[1] = fq[32 + lane];
  for (int ks = 0; ks < NTOK / 32; ++ks) {
    FragU fnext;
    if (ks + 1 < NTOK / 32) {                    // double-buffered B frag
      fnext.q[0] = fq[((ks + 1) << 6) + lane];
      fnext.q[1] = fq[((ks + 1) << 6) + 32 + lane];
    }
    v16bf a = make_a_frag(M + prow * NTOK + ks * 32 + khalf * 8); // p from LDS
    acc = __builtin_amdgcn_wmma_f32_16x16x32_bf16(
            false, a, false, fcur.b, (short)0, acc, false, false);
    if (ks + 1 < NTOK / 32) fcur = fnext;
  }
  float* op = out + (size_t)b * (TD_ + C_ * CD_) + TD_;
  #pragma unroll
  for (int vv = 0; vv < 8; ++vv) {
    int cc = tr * 16 + khalf * 8 + vv;
    op[cc * CD_ + ccol] = 2.0f * acc[vv] - rowsum[cc] * anchors[cc * CD_ + ccol];
  }
}

// ---------------------------------------------------------------------------
// Kernel 3: v_global = t_global @ Wt + bt  (32x768 @ 768x256), WMMA, 1 WG.
// 128 threads = 4 waves; wave -> row-tile (wave&1), 8 col-tiles each.
// ---------------------------------------------------------------------------
__global__ void __launch_bounds__(128)
k_global_proj(const float* __restrict__ x, const float* __restrict__ Wt,
              const float* __restrict__ bt, float* __restrict__ out)
{
  const int tid = threadIdx.x;
  const int wave = tid >> 5, lane = tid & 31;
  const int tr = wave & 1, tcbase = (wave >> 1) * 8;
  const int khalf = lane >> 4;
  const int arow = tr * 16 + (lane & 15);             // batch index
  const float* ar = x + (size_t)arow * N_ * D_;       // token 0 row
  v8f acc[8] = {};
  for (int ks = 0; ks < D_ / 32; ++ks) {
    v16bf a = make_a_frag(ar + ks * 32 + khalf * 8);
    #pragma unroll
    for (int t = 0; t < 8; ++t) {
      int col = (tcbase + t) * 16 + (lane & 15);
      FragU f;
      #pragma unroll
      for (int e = 0; e < 16; e += 2) {
        int k = ks * 32 + kidx(e, khalf);
        f.d[e >> 1] = cvt_pk(Wt[k * TD_ + col], Wt[(k + 1) * TD_ + col]);
      }
      acc[t] = __builtin_amdgcn_wmma_f32_16x16x32_bf16(
                 false, a, false, f.b, (short)0, acc[t], false, false);
    }
  }
  #pragma unroll
  for (int t = 0; t < 8; ++t) {
    int col = (tcbase + t) * 16 + (lane & 15);
    float bias = bt[col];
    #pragma unroll
    for (int vv = 0; vv < 8; ++vv) {
      int row = tr * 16 + khalf * 8 + vv;
      out[(size_t)row * (TD_ + C_ * CD_) + col] = acc[t][vv] + bias;
    }
  }
}

// ---------------------------------------------------------------------------
// Kernel 4: per-batch L2 normalize of the 8448-element output row.
// ---------------------------------------------------------------------------
__global__ void __launch_bounds__(256)
k_l2norm(float* __restrict__ out)
{
  __shared__ float red[8];
  const int b = blockIdx.x, tid = threadIdx.x;
  const int wave = tid >> 5, lane = tid & 31;
  const int ROW = TD_ + C_ * CD_;                     // 8448
  float* p = out + (size_t)b * ROW;
  float ss = 0.0f;
  for (int i = tid; i < ROW; i += 256) { float t = p[i]; ss += t * t; }
  for (int off = 16; off; off >>= 1) ss += __shfl_xor(ss, off, 32);
  if (lane == 0) red[wave] = ss;
  __syncthreads();
  if (tid == 0) {
    float t = 0.0f;
    for (int i = 0; i < 8; ++i) t += red[i];
    red[0] = t;
  }
  __syncthreads();
  const float scale = 1.0f / fmaxf(sqrtf(red[0]), 1e-12f);
  for (int i = tid; i < ROW; i += 256) p[i] *= scale;
}

extern "C" void kernel_launch(void* const* d_in, const int* in_sizes, int n_in,
                              void* d_out, int out_size, void* d_ws, size_t ws_size,
                              hipStream_t stream) {
  const float* x       = (const float*)d_in[0];
  const float* Wf      = (const float*)d_in[1];
  const float* bf      = (const float*)d_in[2];
  const float* Ws      = (const float*)d_in[3];
  const float* bs      = (const float*)d_in[4];
  const float* Wt      = (const float*)d_in[5];
  const float* bt      = (const float*)d_in[6];
  const float* anchors = (const float*)d_in[7];
  const float* dust    = (const float*)d_in[8];
  const float* sharp   = (const float*)d_in[9];
  float* outp = (float*)d_out;

  // ws layout: fbufq = f as bf16 fragments (32*8*32 frags * 1KB = 8 MB),
  //            Mbuf  = f32 Sinkhorn cost (32*64*1024*4 = 8 MB)
  unsigned* fbufq = (unsigned*)d_ws;
  float*    Mbuf  = (float*)((char*)d_ws + (size_t)B_ * 8 * 32 * 1024);

  const size_t lds1 = (size_t)K1_TILES * K1_KSTEPS * 256 * sizeof(unsigned); // 288 KB
  k_fused_gemm<<<dim3((B_ * NTOK) / 128), dim3(256), lds1, stream>>>(
      x, Wf, bf, Ws, bs, sharp, fbufq, Mbuf);

  const size_t lds2 = (size_t)(65 * NTOK + 80 + NTOK + 64) * sizeof(float);  // ~265 KB
  k_sinkhorn_vagg<<<dim3(B_), dim3(1024), lds2, stream>>>(
      Mbuf, fbufq, anchors, dust, outp);

  k_global_proj<<<dim3(1), dim3(128), 0, stream>>>(x, Wt, bt, outp);
  k_l2norm<<<dim3(B_), dim3(256), 0, stream>>>(outp);

  (void)in_sizes; (void)n_in; (void)out_size; (void)ws_size;
}
